// GNN_18159121727555
// MI455X (gfx1250) — matrix operand order repaired and checked by reading, not compiled, not run on hardware
//
#include <hip/hip_runtime.h>
#include <hip/hip_bf16.h>

typedef __attribute__((ext_vector_type(16))) __bf16 v16bf;
typedef __attribute__((ext_vector_type(8)))  float  v8f;

#define GCN_N 100000
#define GCN_E 1600000
#define GCN_F 128
#define GCN_H 64
#define GCN_C 40

// ---------------------------------------------------------------- init / norm
__global__ void gcn_init(float* __restrict__ deg, float* __restrict__ agg1,
                         float* __restrict__ logits) {
  int idx = blockIdx.x * blockDim.x + threadIdx.x;
  if (idx < GCN_N * GCN_H) agg1[idx] = 0.0f;
  if (idx < GCN_N * GCN_C) logits[idx] = 0.0f;
  if (idx < GCN_N) deg[idx] = 1.0f;   // self-loop contributes 1 to every degree
}

__global__ void gcn_degree(const int* __restrict__ dst, float* __restrict__ deg) {
  int e = blockIdx.x * blockDim.x + threadIdx.x;
  if (e < GCN_E) atomicAdd(&deg[dst[e]], 1.0f);
}

__global__ void gcn_rsqrt(float* __restrict__ deg) {
  int i = blockIdx.x * blockDim.x + threadIdx.x;
  if (i < GCN_N) {
    float d = deg[i];
    deg[i] = (d > 0.0f) ? rsqrtf(d) : 0.0f;   // becomes dinv
  }
}

// -------------------------------------------- weight repack: f32 -> bf16 frags
// Fragment layout, contiguous k = 16h + i (same order as A): element i of lane L holds
// W[(kb*32 + (L<16?0:16) + i)*ldw + ct*16 + (L&15)].  One v16bf (32B) per lane.
__global__ void gcn_packW1(const float* __restrict__ W, __bf16* __restrict__ Wb) {
  int idx = blockIdx.x * blockDim.x + threadIdx.x;   // 4ct * 4kb * 32lane * 16
  if (idx >= 4 * 4 * 32 * 16) return;
  int i    = idx & 15;
  int lane = (idx >> 4) & 31;
  int kb   = (idx >> 9) & 3;
  int ct   = idx >> 11;
  int kB   = (lane < 16) ? 0 : 16;
  int n    = (ct << 4) + (lane & 15);
  Wb[idx] = (__bf16)W[(size_t)(kb * 32 + kB + i) * GCN_H + n];
}

__global__ void gcn_packW2(const float* __restrict__ W, __bf16* __restrict__ Wb) {
  int idx = blockIdx.x * blockDim.x + threadIdx.x;   // 3ct * 2kb * 32lane * 16
  if (idx >= 3 * 2 * 32 * 16) return;
  int i    = idx & 15;
  int lane = (idx >> 4) & 31;
  int kb   = (idx >> 9) & 1;
  int ct   = idx >> 10;
  int kB   = (lane < 16) ? 0 : 16;
  int n    = (ct << 4) + (lane & 15);
  Wb[idx] = (n < GCN_C) ? (__bf16)W[(size_t)(kb * 32 + kB + i) * GCN_C + n]
                        : (__bf16)0.0f;
}

// ---------------------------------------------------------------- GEMM1: X@W1
// One wave per 16x16 tile; K=128 -> 4 wmma steps. Exact tiling (6250*4 tiles).
__global__ void gcn_gemm1(const float* __restrict__ X, const v16bf* __restrict__ Wb,
                          float* __restrict__ Hout) {
  const int lane = threadIdx.x & 31;
  const int wave = threadIdx.x >> 5;
  const int tile = blockIdx.x * 8 + wave;
  const int rt = tile >> 2;
  const int ct = tile & 3;
  const int m  = (rt << 4) + (lane & 15);          // A row held by this lane
  const int n  = (ct << 4) + (lane & 15);          // C column held by this lane
  const int kA = (lane < 16) ? 0 : 16;             // contiguous A-fragment K split (k = 16h + i)
  v8f acc = {};
#pragma unroll
  for (int kb = 0; kb < 4; ++kb) {
    v16bf a;
    const float* xr = X + (size_t)m * GCN_F + kb * 32 + kA;
#pragma unroll
    for (int i = 0; i < 8; ++i) { a[i] = (__bf16)xr[i]; a[i + 8] = (__bf16)xr[i + 8]; }
    v16bf b = Wb[(ct * 4 + kb) * 32 + lane];       // one aligned 32B fragment load
    acc = __builtin_amdgcn_wmma_f32_16x16x32_bf16(false, a, false, b,
                                                  (short)0, acc, false, false);
  }
  const int mrow = (rt << 4) + ((lane >> 4) << 3); // C/D: rows j + 8*(lane>=16)
#pragma unroll
  for (int j = 0; j < 8; ++j) Hout[(size_t)(mrow + j) * GCN_H + n] = acc[j];
}

// ---------------------------------------------------------------- GEMM2: H@W2
// K=64 -> 2 wmma steps; C=40 -> store guard on third column tile.
__global__ void gcn_gemm2(const float* __restrict__ Hin, const v16bf* __restrict__ Wb,
                          float* __restrict__ Hout) {
  const int lane = threadIdx.x & 31;
  const int wave = threadIdx.x >> 5;
  const int tile = blockIdx.x * 8 + wave;
  if (tile >= (GCN_N / 16) * 3) return;            // wave-uniform exit
  const int rt = tile / 3;
  const int ct = tile - rt * 3;
  const int m  = (rt << 4) + (lane & 15);
  const int n  = (ct << 4) + (lane & 15);
  const int kA = (lane < 16) ? 0 : 16;
  v8f acc = {};
#pragma unroll
  for (int kb = 0; kb < 2; ++kb) {
    v16bf a;
    const float* xr = Hin + (size_t)m * GCN_H + kb * 32 + kA;
#pragma unroll
    for (int i = 0; i < 8; ++i) { a[i] = (__bf16)xr[i]; a[i + 8] = (__bf16)xr[i + 8]; }
    v16bf b = Wb[(ct * 2 + kb) * 32 + lane];       // pre-zeroed for n>=40
    acc = __builtin_amdgcn_wmma_f32_16x16x32_bf16(false, a, false, b,
                                                  (short)0, acc, false, false);
  }
  if (n < GCN_C) {
    const int mrow = (rt << 4) + ((lane >> 4) << 3);
#pragma unroll
    for (int j = 0; j < 8; ++j) Hout[(size_t)(mrow + j) * GCN_C + n] = acc[j];
  }
}

// ---------------------------------------------------------------- edge scatter
// One wave32 per edge: scalar (SGPR) index/norm loads, lanes cover the feature
// row (coalesced 128B gather), f32 atomics land in L2-resident destination.
// Prefetch the gather row of an edge 64 ahead to hide random HBM latency.
__global__ void gcn_scatter1(const int* __restrict__ src, const int* __restrict__ dst,
                             const float* __restrict__ dinv, const float* __restrict__ h,
                             float* __restrict__ agg) {
  int wid  = (blockIdx.x * blockDim.x + threadIdx.x) >> 5;
  int lane = threadIdx.x & 31;
  if (wid >= GCN_E) return;
  int e = __builtin_amdgcn_readfirstlane(wid);
  if (e + 64 < GCN_E) {
    int sp = src[e + 64];
    __builtin_prefetch(h + (size_t)sp * GCN_H + lane, 0, 1);
  }
  int s = src[e], d = dst[e];
  float w = dinv[s] * dinv[d];
  const float* hr = h + (size_t)s * GCN_H;
  float* ar = agg + (size_t)d * GCN_H;
  atomicAdd(&ar[lane],      hr[lane]      * w);
  atomicAdd(&ar[lane + 32], hr[lane + 32] * w);
}

__global__ void gcn_scatter2(const int* __restrict__ src, const int* __restrict__ dst,
                             const float* __restrict__ dinv, const float* __restrict__ h,
                             float* __restrict__ logits) {
  int wid  = (blockIdx.x * blockDim.x + threadIdx.x) >> 5;
  int lane = threadIdx.x & 31;
  if (wid >= GCN_E) return;
  int e = __builtin_amdgcn_readfirstlane(wid);
  if (e + 64 < GCN_E) {
    int sp = src[e + 64];
    __builtin_prefetch(h + (size_t)sp * GCN_C + lane, 0, 1);
  }
  int s = src[e], d = dst[e];
  float w = dinv[s] * dinv[d];
  const float* hr = h + (size_t)s * GCN_C;
  float* lr = logits + (size_t)d * GCN_C;
  atomicAdd(&lr[lane], hr[lane] * w);
  if (lane < GCN_C - 32)
    atomicAdd(&lr[lane + 32], hr[lane + 32] * w);
}

// ------------------------------------------------- self-loop + bias + ReLU (L1)
__global__ void gcn_bias_relu(const float* __restrict__ h, const float* __restrict__ dinv,
                              const float* __restrict__ bias, float* __restrict__ agg) {
  int idx = blockIdx.x * blockDim.x + threadIdx.x;
  if (idx >= GCN_N * GCN_H) return;
  int i = idx >> 6;
  int f = idx & 63;
  float di = dinv[i];
  float v = agg[idx] + di * di * h[idx] + bias[f];
  agg[idx] = fmaxf(v, 0.0f);
}

// --------------------------- self-loop + bias (L2) then wave32 log_softmax
__global__ void gcn_finalize(const float* __restrict__ h2, const float* __restrict__ dinv,
                             const float* __restrict__ bias, float* __restrict__ logits,
                             float* __restrict__ out) {
  int node = blockIdx.x * 8 + (threadIdx.x >> 5);
  int lane = threadIdx.x & 31;
  if (node >= GCN_N) return;
  float di = dinv[node];
  float w = di * di;
  size_t base = (size_t)node * GCN_C;
  int f0 = lane;                 // always < 40
  int f1 = lane + 32;            // valid for lanes 0..7
  float l0 = logits[base + f0] + w * h2[base + f0] + bias[f0];
  logits[base + f0] = l0;        // features_before_lin (second output half)
  float l1 = -1e30f;
  if (f1 < GCN_C) {
    l1 = logits[base + f1] + w * h2[base + f1] + bias[f1];
    logits[base + f1] = l1;
  }
  float mx = fmaxf(l0, l1);
  for (int off = 16; off > 0; off >>= 1) mx = fmaxf(mx, __shfl_xor(mx, off, 32));
  float s = __expf(l0 - mx) + ((f1 < GCN_C) ? __expf(l1 - mx) : 0.0f);
  for (int off = 16; off > 0; off >>= 1) s += __shfl_xor(s, off, 32);
  float lse = mx + __logf(s);
  out[base + f0] = l0 - lse;
  if (f1 < GCN_C) out[base + f1] = l1 - lse;
}

// ---------------------------------------------------------------------- launch
extern "C" void kernel_launch(void* const* d_in, const int* in_sizes, int n_in,
                              void* d_out, int out_size, void* d_ws, size_t ws_size,
                              hipStream_t stream) {
  const float* x  = (const float*)d_in[0];
  const int*   ei = (const int*)d_in[1];   // (2, E) int32
  const float* W1 = (const float*)d_in[2];
  const float* b1 = (const float*)d_in[3];
  const float* W2 = (const float*)d_in[4];
  const float* b2 = (const float*)d_in[5];
  float* out = (float*)d_out;              // [N*C log_softmax | N*C logits]

  float* ws   = (float*)d_ws;
  float* dinv = ws;                                  // N floats (deg -> dinv)
  float* h1   = ws + 102400;                         // N*H
  float* agg1 = h1 + (size_t)GCN_N * GCN_H;          // N*H (later holds relu'd H)
  float* h2   = agg1 + (size_t)GCN_N * GCN_H;        // N*C
  __bf16* Wb1 = (__bf16*)(h2 + (size_t)GCN_N * GCN_C);   // 8192 bf16, 32B aligned
  __bf16* Wb2 = Wb1 + 4 * 4 * 32 * 16;                   // 3072 bf16
  float* logits = out + (size_t)GCN_N * GCN_C;       // accumulate directly in d_out

  const int* src = ei;
  const int* dst = ei + GCN_E;

  gcn_init    <<<(GCN_N * GCN_H + 255) / 256, 256, 0, stream>>>(dinv, agg1, logits);
  gcn_degree  <<<(GCN_E + 255) / 256,        256, 0, stream>>>(dst, dinv);
  gcn_rsqrt   <<<(GCN_N + 255) / 256,        256, 0, stream>>>(dinv);
  gcn_packW1  <<<(4 * 4 * 32 * 16) / 256,    256, 0, stream>>>(W1, Wb1);
  gcn_packW2  <<<(3 * 2 * 32 * 16) / 256,    256, 0, stream>>>(W2, Wb2);
  gcn_gemm1   <<<(GCN_N / 16) * (GCN_H / 16) / 8, 256, 0, stream>>>(x, (const v16bf*)Wb1, h1);
  gcn_scatter1<<<(GCN_E / 8),                256, 0, stream>>>(src, dst, dinv, h1, agg1);
  gcn_bias_relu<<<(GCN_N * GCN_H + 255) / 256, 256, 0, stream>>>(h1, dinv, b1, agg1);
  gcn_gemm2   <<<((GCN_N / 16) * 3 + 7) / 8, 256, 0, stream>>>(agg1, (const v16bf*)Wb2, h2);
  gcn_scatter2<<<(GCN_E / 8),                256, 0, stream>>>(src, dst, dinv, h2, logits);
  gcn_finalize<<<GCN_N / 8,                  256, 0, stream>>>(h2, dinv, b2, logits, out);
}
